// SparseFeatureCoder_65790309040735
// MI455X (gfx1250) — matrix operand order repaired and checked by reading, not compiled, not run on hardware
//
#include <hip/hip_runtime.h>

// ---------------------------------------------------------------------------
// Problem constants (from reference): D=768, L=16384, M=3072, K=64, B=4096
// ---------------------------------------------------------------------------
#define DIM_D 768
#define DIM_L 16384
#define DIM_M 3072
#define DIM_B 4096
#define K_PER_EX 64
#define K_TOTAL (K_PER_EX * DIM_B)   // 262144

typedef __attribute__((ext_vector_type(16))) __bf16          v16bf;
typedef __attribute__((ext_vector_type(16))) unsigned short  v16us;
typedef __attribute__((ext_vector_type(8)))  float           v8f;
typedef __attribute__((__vector_size__(16))) int             gv4i;

// CDNA5 async global->LDS path (ASYNCcnt), guarded so either toolchain builds.
#if defined(__has_builtin)
#  if __has_builtin(__builtin_amdgcn_global_load_async_to_lds_b128)
#    define HAVE_ASYNC_LDS 1
#  endif
#  if __has_builtin(__builtin_amdgcn_s_wait_asynccnt)
#    define HAVE_WAIT_ASYNC 1
#  endif
#endif

// Exact parameter types per hipcc diagnostic: AS1 (global) / AS3 (LDS) v4i*.
#define ASG4(p) ((__attribute__((address_space(1))) gv4i*)(p))
#define ASL4(p) ((__attribute__((address_space(3))) gv4i*)(p))

static __device__ __forceinline__ void wait_async_zero() {
#if defined(HAVE_WAIT_ASYNC)
    __builtin_amdgcn_s_wait_asynccnt(0);
#elif defined(HAVE_ASYNC_LDS)
    asm volatile("s_wait_asynccnt 0x0" ::: "memory");
#endif
}

static __device__ __forceinline__ unsigned short f2bf(float f) {
    unsigned int u = __builtin_bit_cast(unsigned int, f);
    u += 0x7fffu + ((u >> 16) & 1u);          // round-to-nearest-even
    return (unsigned short)(u >> 16);
}

// Monotonic float -> uint key (ascending); top-k == largest keys.
static __device__ __forceinline__ unsigned int fkey(float f) {
    unsigned int u = __builtin_bit_cast(unsigned int, f);
    return (u & 0x80000000u) ? ~u : (u | 0x80000000u);
}

// ---------------------------------------------------------------------------
// fp32 -> bf16 conversion (weights / activations)
// ---------------------------------------------------------------------------
__global__ void cvt_f32_bf16(const float* __restrict__ in,
                             unsigned short* __restrict__ out, size_t n) {
    size_t i = (size_t)blockIdx.x * blockDim.x + threadIdx.x;
    size_t stride = (size_t)gridDim.x * blockDim.x;
    for (; i < n; i += stride) out[i] = f2bf(in[i]);
}

// ---------------------------------------------------------------------------
// bf16 WMMA GEMM:  C[M,N] = A[M,K] * W[K,N]  (+bias) (+relu)
// 256 threads = 8 waves; block tile 128x128, K-tile 32, double-buffered LDS,
// software-pipelined (async global->LDS for A when available).
// Wave grid 2(M) x 4(N): each wave -> 64x32 -> 4x2 accumulators of 16x16.
// ---------------------------------------------------------------------------
#define BMT 128
#define BNT 128
#define BKT 32

template <bool HAS_BIAS, bool RELU, bool OUT_BF16>
__global__ __launch_bounds__(256)
void gemm_bf16(const unsigned short* __restrict__ A,   // M x K, bf16 bits
               const unsigned short* __restrict__ W,   // K x N, bf16 bits
               const float* __restrict__ bias,         // N (optional)
               void* __restrict__ Cout,                // M x N
               int Kdim, int Ndim) {
    __shared__ __align__(32) unsigned short As[2][BMT * BKT];  // [m][k]
    __shared__ __align__(32) unsigned short Bs[2][BNT * BKT];  // [n][k] (transposed)

    const int tid  = threadIdx.x;
    const int lane = tid & 31;
    const int wave = tid >> 5;       // 0..7
    const int wm   = wave >> 2;      // 0..1 -> 64 rows
    const int wn   = wave & 3;       // 0..3 -> 32 cols
    const int half = lane >> 4;      // 0/1
    const int l16  = lane & 15;
    const int bm   = blockIdx.y * BMT;
    const int bn   = blockIdx.x * BNT;

    // Per-thread staging coordinates (2 uint4 chunks each for A and B tiles).
    int ar[2], ac[2], bk[2], bn8[2];
    #pragma unroll
    for (int i = 0; i < 2; ++i) {
        int g  = tid + i * 256;
        ar[i]  = g >> 2;           // A row 0..127
        ac[i]  = (g & 3) * 8;      // A col chunk
        bk[i]  = g >> 4;           // W k-row 0..31
        bn8[i] = (g & 15) * 8;     // W n chunk
    }

    v8f acc[4][2] = {};
    uint4 aregs[2], bregs[2];
    (void)aregs;

    // ---- staging helpers ----
    auto issueA = [&](int buf, int k0) {
        #pragma unroll
        for (int i = 0; i < 2; ++i) {
#if defined(HAVE_ASYNC_LDS)
            __builtin_amdgcn_global_load_async_to_lds_b128(
                ASG4(&A[(size_t)(bm + ar[i]) * Kdim + k0 + ac[i]]),
                ASL4(&As[buf][ar[i] * BKT + ac[i]]), 0, 0);
#else
            aregs[i] = *reinterpret_cast<const uint4*>(
                &A[(size_t)(bm + ar[i]) * Kdim + k0 + ac[i]]);
#endif
        }
    };
    auto commitA = [&](int buf) {
#if !defined(HAVE_ASYNC_LDS)
        #pragma unroll
        for (int i = 0; i < 2; ++i)
            *reinterpret_cast<uint4*>(&As[buf][ar[i] * BKT + ac[i]]) = aregs[i];
#else
        (void)buf;
#endif
    };
    auto issueB = [&](int k0) {
        #pragma unroll
        for (int i = 0; i < 2; ++i)
            bregs[i] = *reinterpret_cast<const uint4*>(
                &W[(size_t)(k0 + bk[i]) * Ndim + bn + bn8[i]]);
    };
    auto commitB = [&](int buf) {
        #pragma unroll
        for (int i = 0; i < 2; ++i) {
            const unsigned short* p =
                reinterpret_cast<const unsigned short*>(&bregs[i]);
            #pragma unroll
            for (int e = 0; e < 8; ++e)
                Bs[buf][(bn8[i] + e) * BKT + bk[i]] = p[e];
        }
    };

    // ---- compute one K-tile from LDS buffer `buf` ----
    auto compute = [&](int buf) {
        v16bf afrag[4];
        #pragma unroll
        for (int mt = 0; mt < 4; ++mt) {
            int m = wm * 64 + mt * 16 + l16;
            const unsigned short* arow = &As[buf][m * BKT];
            v16us au;
            #pragma unroll
            for (int v = 0; v < 8; ++v) {
                int k = (half << 4) + (v << 1);
                unsigned int u =
                    *reinterpret_cast<const unsigned int*>(&arow[k]);
                au[2 * v]     = (unsigned short)(u & 0xffffu);
                au[2 * v + 1] = (unsigned short)(u >> 16);
            }
            afrag[mt] = __builtin_bit_cast(v16bf, au);
        }
        #pragma unroll
        for (int nt = 0; nt < 2; ++nt) {
            int n = wn * 32 + nt * 16 + l16;
            v16bf bfrag = *reinterpret_cast<const v16bf*>(
                &Bs[buf][n * BKT + half * 16]);
            #pragma unroll
            for (int mt = 0; mt < 4; ++mt)
                acc[mt][nt] = __builtin_amdgcn_wmma_f32_16x16x32_bf16(
                    false, afrag[mt], false, bfrag, (short)0, acc[mt][nt],
                    false, false);
        }
    };

    // ---- pipelined main loop: one barrier per K-step ----
    issueA(0, 0);
    issueB(0);
    commitA(0);
    commitB(0);
    wait_async_zero();
    __syncthreads();

    int cur = 0;
    for (int k0 = BKT; k0 < Kdim; k0 += BKT) {
        issueA(cur ^ 1, k0);     // async HW copy (or loads into regs)
        issueB(k0);              // loads into regs
        compute(cur);            // 8 WMMAs overlap the in-flight loads
        commitA(cur ^ 1);
        commitB(cur ^ 1);
        wait_async_zero();
        __syncthreads();
        cur ^= 1;
    }
    compute(cur);

    // ---- epilogue ----
    #pragma unroll
    for (int nt = 0; nt < 2; ++nt) {
        int gn = bn + wn * 32 + nt * 16 + l16;
        float bv = HAS_BIAS ? bias[gn] : 0.0f;
        #pragma unroll
        for (int mt = 0; mt < 4; ++mt) {
            #pragma unroll
            for (int r = 0; r < 8; ++r) {
                int gm = bm + wm * 64 + mt * 16 + half * 8 + r;
                float val = acc[mt][nt][r] + bv;
                if (RELU) val = fmaxf(val, 0.0f);
                if (OUT_BF16)
                    ((unsigned short*)Cout)[(size_t)gm * Ndim + gn] = f2bf(val);
                else
                    ((float*)Cout)[(size_t)gm * Ndim + gn] = val;
            }
        }
    }
}

// ---------------------------------------------------------------------------
// BatchTopK via 4-round byte radix select on float keys.
// sel[0]=prefix, sel[1]=k_remaining, sel[2]=tie counter
// ---------------------------------------------------------------------------
__global__ void sel_init(unsigned int* sel, unsigned int ktotal) {
    if (threadIdx.x == 0) { sel[0] = 0u; sel[1] = ktotal; sel[2] = 0u; }
}

__global__ void zero_hist(unsigned int* h) { h[threadIdx.x] = 0u; }

__global__ __launch_bounds__(256)
void radix_hist(const float* __restrict__ data, size_t n,
                const unsigned int* __restrict__ sel,
                unsigned int* __restrict__ ghist,
                int shift, unsigned int himask) {
    __shared__ unsigned int lh[256];
    lh[threadIdx.x] = 0u;
    __syncthreads();
    unsigned int prefix = sel[0];
    size_t i = (size_t)blockIdx.x * blockDim.x + threadIdx.x;
    size_t stride = (size_t)gridDim.x * blockDim.x;
    for (; i < n; i += stride) {
        unsigned int k = fkey(data[i]);
        if ((k & himask) == prefix)
            atomicAdd(&lh[(k >> shift) & 0xffu], 1u);
    }
    __syncthreads();
    unsigned int c = lh[threadIdx.x];
    if (c) atomicAdd(&ghist[threadIdx.x], c);
}

__global__ void radix_pick(unsigned int* sel, const unsigned int* ghist,
                           int shift) {
    if (threadIdx.x != 0) return;
    unsigned int k = sel[1];
    unsigned int cum = 0u;
    int b;
    for (b = 255; b > 0; --b) {
        unsigned int c = ghist[b];
        if (cum + c >= k) break;
        cum += c;
    }
    sel[0] |= ((unsigned int)b) << shift;
    sel[1] = k - cum;
}

// Keep values above threshold (+ limited ties), write sparsified bf16.
__global__ __launch_bounds__(256)
void topk_apply(const float* __restrict__ data, size_t n,
                unsigned int* __restrict__ sel,
                unsigned short* __restrict__ out_bf) {
    const unsigned int thr  = sel[0];
    const unsigned int ties = sel[1];
    size_t i = (size_t)blockIdx.x * blockDim.x + threadIdx.x;
    size_t stride = (size_t)gridDim.x * blockDim.x;
    for (; i < n; i += stride) {
        float x = data[i];
        unsigned int k = fkey(x);
        float r = 0.0f;
        if (k > thr) r = x;
        else if (k == thr) {
            if (atomicAdd(&sel[2], 1u) < ties) r = x;
        }
        out_bf[i] = f2bf(r);
    }
}

// ---------------------------------------------------------------------------
// Host side
// ---------------------------------------------------------------------------
static void run_topk(const float* data, size_t n, unsigned int* sel,
                     unsigned int* ghist, unsigned short* out_bf,
                     hipStream_t stream) {
    sel_init<<<1, 32, 0, stream>>>(sel, K_TOTAL);
    const int shifts[4]         = {24, 16, 8, 0};
    const unsigned int masks[4] = {0x00000000u, 0xFF000000u,
                                   0xFFFF0000u, 0xFFFFFF00u};
    for (int r = 0; r < 4; ++r) {
        zero_hist<<<1, 256, 0, stream>>>(ghist);
        radix_hist<<<1024, 256, 0, stream>>>(data, n, sel, ghist,
                                             shifts[r], masks[r]);
        radix_pick<<<1, 32, 0, stream>>>(sel, ghist, shifts[r]);
    }
    topk_apply<<<2048, 256, 0, stream>>>(data, n, sel, out_bf);
}

extern "C" void kernel_launch(void* const* d_in, const int* in_sizes, int n_in,
                              void* d_out, int out_size, void* d_ws,
                              size_t ws_size, hipStream_t stream) {
    (void)in_sizes; (void)n_in; (void)out_size; (void)ws_size;

    const float* acts = (const float*)d_in[0];   // B x D
    const float* Wes  = (const float*)d_in[1];   // D x L
    const float* Wem  = (const float*)d_in[2];   // L x M
    const float* bem  = (const float*)d_in[3];   // M
    const float* Wdm  = (const float*)d_in[4];   // M x L
    const float* bdl  = (const float*)d_in[5];   // L
    const float* Wds  = (const float*)d_in[6];   // L x D
    float* out = (float*)d_out;                  // B x D

    // ---- carve workspace ----
    char* p = (char*)d_ws;
    auto take = [&](size_t bytes) -> void* {
        void* r = (void*)p;
        p += (bytes + 255) & ~(size_t)255;
        return r;
    };
    unsigned short* bf_acts = (unsigned short*)take((size_t)DIM_B * DIM_D * 2);
    unsigned short* bf_Wes  = (unsigned short*)take((size_t)DIM_D * DIM_L * 2);
    unsigned short* bf_Wem  = (unsigned short*)take((size_t)DIM_L * DIM_M * 2);
    unsigned short* bf_Wdm  = (unsigned short*)take((size_t)DIM_M * DIM_L * 2);
    unsigned short* bf_Wds  = (unsigned short*)take((size_t)DIM_L * DIM_D * 2);
    float*          preBL   = (float*)take((size_t)DIM_B * DIM_L * 4); // reused
    unsigned short* latBL   = (unsigned short*)take((size_t)DIM_B * DIM_L * 2);
    unsigned short* hidBM   = (unsigned short*)take((size_t)DIM_B * DIM_M * 2);
    unsigned int*   ghist   = (unsigned int*)take(256 * 4);
    unsigned int*   sel     = (unsigned int*)take(64);

    // ---- fp32 -> bf16 conversions ----
    cvt_f32_bf16<<<2048, 256, 0, stream>>>(acts, bf_acts, (size_t)DIM_B * DIM_D);
    cvt_f32_bf16<<<2048, 256, 0, stream>>>(Wes,  bf_Wes,  (size_t)DIM_D * DIM_L);
    cvt_f32_bf16<<<4096, 256, 0, stream>>>(Wem,  bf_Wem,  (size_t)DIM_L * DIM_M);
    cvt_f32_bf16<<<4096, 256, 0, stream>>>(Wdm,  bf_Wdm,  (size_t)DIM_M * DIM_L);
    cvt_f32_bf16<<<2048, 256, 0, stream>>>(Wds,  bf_Wds,  (size_t)DIM_L * DIM_D);

    // ---- GEMM1: pre = acts @ W_enc_sparse  (4096x16384, K=768) ----
    gemm_bf16<false, false, false>
        <<<dim3(DIM_L / BNT, DIM_B / BMT), 256, 0, stream>>>(
            bf_acts, bf_Wes, nullptr, preBL, DIM_D, DIM_L);

    // ---- TopK #1 -> bf16 latents ----
    run_topk(preBL, (size_t)DIM_B * DIM_L, sel, ghist, latBL, stream);

    // ---- GEMM2: hid = relu(lat @ W_enc_mlp + b)  (4096x3072, K=16384) ----
    gemm_bf16<true, true, true>
        <<<dim3(DIM_M / BNT, DIM_B / BMT), 256, 0, stream>>>(
            latBL, bf_Wem, bem, hidBM, DIM_L, DIM_M);

    // ---- GEMM3: mlp_out = hid @ W_dec_mlp + b  (4096x16384, K=3072) ----
    gemm_bf16<true, false, false>
        <<<dim3(DIM_L / BNT, DIM_B / BMT), 256, 0, stream>>>(
            hidBM, bf_Wdm, bdl, preBL, DIM_M, DIM_L);

    // ---- TopK #2 -> bf16 latents_out ----
    run_topk(preBL, (size_t)DIM_B * DIM_L, sel, ghist, latBL, stream);

    // ---- GEMM4: out = lat_out @ W_dec_sparse  (4096x768, K=16384) ----
    gemm_bf16<false, false, false>
        <<<dim3(DIM_D / BNT, DIM_B / BMT), 256, 0, stream>>>(
            latBL, bf_Wds, nullptr, out, DIM_L, DIM_D);
}
